// knn_seg_model_2894807957816
// MI455X (gfx1250) — matrix operand order repaired and checked by reading, not compiled, not run on hardware
//
#include <hip/hip_runtime.h>
#include <hip/hip_bf16.h>

// ---------------------------------------------------------------------------
// DGCNN part-segmentation forward for MI455X (gfx1250), wave32 WMMA.
// Conv1x1 GEMMs use v_wmma_f32_16x16x32_bf16 (bf16 A/B, fp32 accumulate).
//
// Activation layout "pack16": tensor [C][P] stored as [C/16][P][16] so a
// B-matrix WMMA fragment (16 K-values at one column) is 32 contiguous bytes
// -> 2x global_load_b128 per fragment; the GEMM epilogue writes one b128 per
// tile. The K loop is software-pipelined (double-buffered fragments) so the
// loads for chunk k+1 are in flight while chunk k's WMMAs execute.
// ---------------------------------------------------------------------------

typedef __bf16 bf16_t;
typedef __attribute__((ext_vector_type(16))) __bf16 v16bf;
typedef __attribute__((ext_vector_type(8)))  __bf16 v8bf;
typedef __attribute__((ext_vector_type(8)))  float  v8f;

#define BATCH 8
#define NPTS  2048
#define KNN   10
#define P1    (BATCH * NPTS)        // 16384
#define PK    (P1 * KNN)            // 163840
#define NEG   0.2f

static __device__ __forceinline__ float lrelu(float x) {
    return x > 0.0f ? x : NEG * x;
}

// pack16 element address: channel c, column p, column-count P
static __device__ __forceinline__ size_t pidx(int c, size_t p, size_t P) {
    return ((size_t)(c >> 4) * P + p) * 16 + (c & 15);
}

// ---------------------------------------------------------------------------
// Fused GEMM: Y[co][p] = act(scale[co] * sum_ci W[co][ci]*X[ci][p] + bias[co])
// W : bf16 [Co][CiPad] row-major (CiPad % 32 == 0, zero padded)
// X : bf16 pack16 [CiPad/16][P][16]   (P % 256 == 0)
// Y : bf16 pack16 [Co/16][P][16]
// Block = 128 threads = 4 waves; each wave computes a 16(Co) x 64(P) strip.
// ---------------------------------------------------------------------------
__global__ void k_gemm_bn_act(const bf16_t* __restrict__ W,
                              const bf16_t* __restrict__ X,
                              const float*  __restrict__ scale,
                              const float*  __restrict__ bias,
                              bf16_t* __restrict__ Y,
                              int Co, int CiPad, int P, int act)
{
    const int lane = threadIdx.x & 31;
    const int wave = threadIdx.x >> 5;
    const int co0  = blockIdx.y * 16;
    const int p0   = blockIdx.x * 256 + wave * 64;
    const int m    = lane & 15;     // A row / B,D column within tile
    const int half = lane >> 4;     // K-half selector (contiguous K order used by this code)

    const bf16_t* wrow = W + (size_t)(co0 + m) * CiPad;
    const int nk = CiPad >> 5;      // number of 32-wide K chunks

    // A fragment (16x32 bf16): element e <-> K = k0 + half*16 + e
    auto loadA = [&](int k0) -> v16bf {
        union { v16bf v; v8bf h[2]; } u;
        u.h[0] = *(const v8bf*)(wrow + k0 + half * 16);
        u.h[1] = *(const v8bf*)(wrow + k0 + half * 16 + 8);
        return u.v;
    };
    // B fragment: element e <-> K = k0 + half*16 + e  == pack16 block
    // (k0/16 + half), 16 contiguous bf16 at column pc.
    auto loadB = [&](int k0, int t) -> v16bf {
        const size_t blk = (size_t)((k0 >> 4) + half);
        const size_t pc  = (size_t)(p0 + t * 16 + m);
        return *(const v16bf*)(X + (blk * P + pc) * 16);
    };

    v8f acc[4] = {};

    // ---- software pipeline: prologue ----
    v16bf a_cur = loadA(0);
    v16bf b_cur[4];
#pragma unroll
    for (int t = 0; t < 4; ++t) b_cur[t] = loadB(0, t);

    for (int kc = 0; kc < nk; ++kc) {
        // Clamped next-chunk index keeps the loop branchless; the final
        // iteration redundantly reloads the last chunk (discarded).
        const int knext = (kc + 1 < nk ? kc + 1 : nk - 1) * 32;
        const int kpref = (kc + 2 < nk ? kc + 2 : nk - 1) * 32;

        // distance-2 prefetch -> global_prefetch_b8
        __builtin_prefetch(wrow + kpref, 0, 1);
        __builtin_prefetch(X + ((size_t)((kpref >> 4) + half) * P + p0 + m) * 16, 0, 1);

        // issue next chunk's loads before consuming current fragments
        v16bf a_nxt = loadA(knext);
        v16bf b_nxt[4];
#pragma unroll
        for (int t = 0; t < 4; ++t) b_nxt[t] = loadB(knext, t);

#pragma unroll
        for (int t = 0; t < 4; ++t)
            acc[t] = __builtin_amdgcn_wmma_f32_16x16x32_bf16(
                false, a_cur, false, b_cur[t], (short)0, acc[t], false, false);

        a_cur = a_nxt;
#pragma unroll
        for (int t = 0; t < 4; ++t) b_cur[t] = b_nxt[t];
    }

    // Epilogue. D element (M = r + 8*half, N = m) lives in acc[t][r].
    // A lane's 8 rows are channels co0 + 8*half + r -> contiguous in pack16.
#pragma unroll
    for (int t = 0; t < 4; ++t) {
        const int p = p0 + t * 16 + m;
        v8bf o;
#pragma unroll
        for (int r = 0; r < 8; ++r) {
            const int co = co0 + 8 * half + r;
            float y = acc[t][r] * scale[co] + bias[co];
            if (act) y = lrelu(y);
            o[r] = (bf16_t)y;
        }
        *(v8bf*)(Y + ((size_t)(co0 >> 4) * P + p) * 16 + 8 * half) = o;
    }
}

// ---------------------------------------------------------------------------
// Weight convert fp32 [Co][Ci] -> bf16 [Co][CiPad] (zero padded).
// ---------------------------------------------------------------------------
__global__ void k_cvt_w(const float* __restrict__ src, bf16_t* __restrict__ dst,
                        int Co, int Ci, int CiPad)
{
    const int i = blockIdx.x * blockDim.x + threadIdx.x;
    if (i >= Co * CiPad) return;
    const int co = i / CiPad, ci = i - co * CiPad;
    dst[i] = (bf16_t)((ci < Ci) ? src[co * Ci + ci] : 0.0f);
}

// ---------------------------------------------------------------------------
// Register-resident top-10 insertion (sorted ascending, fully unrolled).
// ---------------------------------------------------------------------------
static __device__ __forceinline__ void topk_insert(float d, int j, float* bd, int* bi)
{
    if (d < bd[KNN - 1]) {
        bd[KNN - 1] = d; bi[KNN - 1] = j;
#pragma unroll
        for (int s = KNN - 1; s > 0; --s) {
            if (bd[s] < bd[s - 1]) {
                float td = bd[s]; bd[s] = bd[s - 1]; bd[s - 1] = td;
                int   ti = bi[s]; bi[s] = bi[s - 1]; bi[s - 1] = ti;
            }
        }
    }
}

// KNN on raw xyz (B,N,3). idx: [B*N][KNN] (per-batch neighbor indices).
__global__ void k_knn3(const float* __restrict__ pts, int* __restrict__ idx)
{
    const int gid = blockIdx.x * blockDim.x + threadIdx.x;
    if (gid >= P1) return;
    const int b = gid / NPTS, n = gid - b * NPTS;
    const float* base = pts + (size_t)b * NPTS * 3;
    const float qx = base[n * 3 + 0], qy = base[n * 3 + 1], qz = base[n * 3 + 2];

    float bd[KNN]; int bi[KNN];
#pragma unroll
    for (int k = 0; k < KNN; ++k) { bd[k] = 3.0e38f; bi[k] = 0; }

    for (int mth = 0; mth < NPTS; ++mth) {
        if (mth == n) continue;
        const float dx = base[mth * 3 + 0] - qx;
        const float dy = base[mth * 3 + 1] - qy;
        const float dz = base[mth * 3 + 2] - qz;
        topk_insert(dx * dx + dy * dy + dz * dz, mth, bd, bi);
    }
#pragma unroll
    for (int k = 0; k < KNN; ++k) idx[(size_t)gid * KNN + k] = bi[k];
}

// KNN on 64-ch pack16 features at channel offset rowoff (multiple of 16).
// Block = 64 threads = 64 query points of one batch; LDS-tiled scan.
__global__ void k_knn_feat(const bf16_t* __restrict__ F, int rowoff, int* __restrict__ idx)
{
    __shared__ float q[64][65];
    __shared__ float mmv[64][65];
    const int b  = blockIdx.y;
    const int n0 = blockIdx.x * 64;
    const int t  = threadIdx.x;
    const int n  = n0 + t;
    const int rb = rowoff >> 4;

#pragma unroll
    for (int bb = 0; bb < 4; ++bb) {
        v16bf v = *(const v16bf*)(F + ((size_t)(rb + bb) * P1 + (size_t)b * NPTS + n) * 16);
#pragma unroll
        for (int e = 0; e < 16; ++e) q[t][bb * 16 + e] = (float)v[e];
    }

    float bd[KNN]; int bi[KNN];
#pragma unroll
    for (int k = 0; k < KNN; ++k) { bd[k] = 3.0e38f; bi[k] = 0; }

    for (int m0 = 0; m0 < NPTS; m0 += 64) {
        __syncthreads();
#pragma unroll
        for (int bb = 0; bb < 4; ++bb) {
            v16bf v = *(const v16bf*)(F + ((size_t)(rb + bb) * P1 + (size_t)b * NPTS + m0 + t) * 16);
#pragma unroll
            for (int e = 0; e < 16; ++e) mmv[t][bb * 16 + e] = (float)v[e];
        }
        __syncthreads();
        for (int mj = 0; mj < 64; ++mj) {
            const int mth = m0 + mj;
            if (mth == n) continue;
            float d = 0.0f;
#pragma unroll 8
            for (int c = 0; c < 64; ++c) {
                const float df = q[t][c] - mmv[mj][c];
                d += df * df;
            }
            topk_insert(d, mth, bd, bi);
        }
    }
#pragma unroll
    for (int k = 0; k < KNN; ++k) idx[((size_t)b * NPTS + n) * KNN + k] = bi[k];
}

// Edge features from xyz -> X pack16 [32ch][PK]: ch0..2 = feat-center,
// ch3..5 = center, ch6..31 = 0.
__global__ void k_edge3(const float* __restrict__ pts, const int* __restrict__ idx,
                        bf16_t* __restrict__ X)
{
    const int p = blockIdx.x * blockDim.x + threadIdx.x;
    if (p >= PK) return;
    const int bn = p / KNN;
    const int b = bn / NPTS, n = bn - b * NPTS;
    const int j = idx[p];
    const float* base = pts + (size_t)b * NPTS * 3;

    v16bf blk0 = {};
#pragma unroll
    for (int c = 0; c < 3; ++c) {
        const float ctr = base[n * 3 + c];
        const float ft  = base[j * 3 + c];
        blk0[c]     = (bf16_t)(ft - ctr);
        blk0[3 + c] = (bf16_t)ctr;
    }
    v16bf blk1 = {};
    *(v16bf*)(X + ((size_t)0 * PK + p) * 16) = blk0;
    *(v16bf*)(X + ((size_t)1 * PK + p) * 16) = blk1;
}

// Edge features from 64-ch pack16 features -> X pack16 [128ch][PK].
__global__ void k_edge_feat(const bf16_t* __restrict__ F, int rowoff,
                            const int* __restrict__ idx, bf16_t* __restrict__ X)
{
    const int p = blockIdx.x * blockDim.x + threadIdx.x;
    if (p >= PK) return;
    const int bn = p / KNN;
    const int b = bn / NPTS, n = bn - b * NPTS;
    const int j = idx[p];
    const int rb = rowoff >> 4;
    const size_t bcol = (size_t)b * NPTS;

#pragma unroll
    for (int bb = 0; bb < 4; ++bb) {
        v16bf ctr = *(const v16bf*)(F + ((size_t)(rb + bb) * P1 + bcol + n) * 16);
        v16bf ft  = *(const v16bf*)(F + ((size_t)(rb + bb) * P1 + bcol + j) * 16);
        v16bf dv;
#pragma unroll
        for (int e = 0; e < 16; ++e) dv[e] = (bf16_t)((float)ft[e] - (float)ctr[e]);
        *(v16bf*)(X + ((size_t)bb * PK + p) * 16)       = dv;
        *(v16bf*)(X + ((size_t)(4 + bb) * PK + p) * 16) = ctr;
    }
}

// Max over neighbor dim: Y pack16 [64][PK] -> O pack16 rows rowoff.. over P1.
__global__ void k_maxk(const bf16_t* __restrict__ Y, bf16_t* __restrict__ O, int rowoff)
{
    const int i = blockIdx.x * blockDim.x + threadIdx.x;
    if (i >= 64 * P1) return;
    const int co = i / P1, p = i - co * P1;
    float mx = -3.0e38f;
#pragma unroll
    for (int k = 0; k < KNN; ++k)
        mx = fmaxf(mx, (float)Y[pidx(co, (size_t)p * KNN + k, PK)]);
    O[pidx(rowoff + co, p, P1)] = (bf16_t)mx;
}

// Max over points per batch: Y pack16 [1024][P1] -> out[B][1024] (fp32).
__global__ void k_maxn(const bf16_t* __restrict__ Y, float* __restrict__ out)
{
    const int i = blockIdx.x * blockDim.x + threadIdx.x;
    if (i >= BATCH * 1024) return;
    const int b = i / 1024, co = i - b * 1024;
    float mx = -3.0e38f;
    for (int n = 0; n < NPTS; ++n)
        mx = fmaxf(mx, (float)Y[pidx(co, (size_t)b * NPTS + n, P1)]);
    out[i] = mx;
}

// Label branch: catvet[b][co] = lrelu(s7[co]*(w7[co]·label[b]) + b7[co]).
__global__ void k_label(const float* __restrict__ label, const float* __restrict__ w7,
                        const float* __restrict__ s7, const float* __restrict__ b7,
                        float* __restrict__ catvet)
{
    const int i = blockIdx.x * blockDim.x + threadIdx.x;
    if (i >= BATCH * 64) return;
    const int b = i / 64, co = i - b * 64;
    float acc = 0.0f;
#pragma unroll
    for (int c = 0; c < 16; ++c) acc += w7[co * 16 + c] * label[b * 16 + c];
    catvet[i] = lrelu(acc * s7[co] + b7[co]);
}

// Broadcast global features into channels 192..1279 of X1280 (pack16).
__global__ void k_glob(const float* __restrict__ out4max, const float* __restrict__ catvet,
                       bf16_t* __restrict__ X1280)
{
    const int i = blockIdx.x * blockDim.x + threadIdx.x;
    if (i >= 1088 * P1) return;
    const int row = i / P1, p = i - row * P1;
    const int b = p / NPTS;
    const float v = (row < 1024) ? out4max[b * 1024 + row] : catvet[b * 64 + (row - 1024)];
    X1280[pidx(192 + row, p, P1)] = (bf16_t)v;
}

// Final 128 -> 6 head (no BN/act), fp32 output (B, N, 6).
__global__ void k_final(const bf16_t* __restrict__ Y, const float* __restrict__ wp4,
                        float* __restrict__ out)
{
    const int p = blockIdx.x * blockDim.x + threadIdx.x;
    if (p >= P1) return;
    float acc[6] = {};
#pragma unroll
    for (int bb = 0; bb < 8; ++bb) {
        v16bf xv = *(const v16bf*)(Y + ((size_t)bb * P1 + p) * 16);
#pragma unroll
        for (int e = 0; e < 16; ++e) {
            const float xe = (float)xv[e];
#pragma unroll
            for (int o = 0; o < 6; ++o)
                acc[o] += wp4[o * 128 + bb * 16 + e] * xe;
        }
    }
#pragma unroll
    for (int o = 0; o < 6; ++o) out[(size_t)p * 6 + o] = acc[o];
}

// ---------------------------------------------------------------------------
// Host-side orchestration.
// ---------------------------------------------------------------------------
static inline size_t alignup(size_t x) { return (x + 255) & ~(size_t)255; }

extern "C" void kernel_launch(void* const* d_in, const int* in_sizes, int n_in,
                              void* d_out, int out_size, void* d_ws, size_t ws_size,
                              hipStream_t stream)
{
    const float* points = (const float*)d_in[0];
    const float* label  = (const float*)d_in[1];
    const float* w1  = (const float*)d_in[2];  const float* s1  = (const float*)d_in[3];  const float* b1  = (const float*)d_in[4];
    const float* w2  = (const float*)d_in[5];  const float* s2  = (const float*)d_in[6];  const float* b2  = (const float*)d_in[7];
    const float* w3  = (const float*)d_in[8];  const float* s3  = (const float*)d_in[9];  const float* b3  = (const float*)d_in[10];
    const float* w4  = (const float*)d_in[11]; const float* s4  = (const float*)d_in[12]; const float* b4  = (const float*)d_in[13];
    const float* w5  = (const float*)d_in[14]; const float* s5  = (const float*)d_in[15]; const float* b5  = (const float*)d_in[16];
    const float* w6  = (const float*)d_in[17]; const float* s6  = (const float*)d_in[18]; const float* b6  = (const float*)d_in[19];
    const float* w7  = (const float*)d_in[20]; const float* s7  = (const float*)d_in[21]; const float* b7  = (const float*)d_in[22];
    const float* wp1 = (const float*)d_in[23]; const float* sp1 = (const float*)d_in[24]; const float* bp1 = (const float*)d_in[25];
    const float* wp2 = (const float*)d_in[26]; const float* sp2 = (const float*)d_in[27]; const float* bp2 = (const float*)d_in[28];
    const float* wp3 = (const float*)d_in[29]; const float* sp3 = (const float*)d_in[30]; const float* bp3 = (const float*)d_in[31];
    const float* wp4 = (const float*)d_in[32];
    float* out = (float*)d_out;

    // ---- workspace carve ----
    char* ws = (char*)d_ws;
    size_t off = 0;
    auto carve = [&](size_t bytes) { char* p = ws + off; off += alignup(bytes); return p; };

    int*    idx      = (int*)   carve(sizeof(int) * PK);
    bf16_t* e_buf    = (bf16_t*)carve(sizeof(bf16_t) * 128 * (size_t)PK);   // pack16 [8][PK][16]
    bf16_t* h_buf    = (bf16_t*)carve(sizeof(bf16_t) * 64  * (size_t)PK);   // pack16 [4][PK][16]
    bf16_t* X1280    = (bf16_t*)carve(sizeof(bf16_t) * 1280 * (size_t)P1);  // pack16 [80][P1][16]
    bf16_t* out4full = (bf16_t*)carve(sizeof(bf16_t) * 1024 * (size_t)P1);
    bf16_t* y256a    = (bf16_t*)carve(sizeof(bf16_t) * 256 * (size_t)P1);
    bf16_t* y256b    = (bf16_t*)carve(sizeof(bf16_t) * 256 * (size_t)P1);
    bf16_t* y128     = (bf16_t*)carve(sizeof(bf16_t) * 128 * (size_t)P1);
    float*  out4max  = (float*) carve(sizeof(float) * BATCH * 1024);
    float*  catvet   = (float*) carve(sizeof(float) * BATCH * 64);
    bf16_t* w1b  = (bf16_t*)carve(sizeof(bf16_t) * 64 * 32);
    bf16_t* w2b  = (bf16_t*)carve(sizeof(bf16_t) * 64 * 64);
    bf16_t* w3b  = (bf16_t*)carve(sizeof(bf16_t) * 64 * 128);
    bf16_t* w4b  = (bf16_t*)carve(sizeof(bf16_t) * 64 * 64);
    bf16_t* w5b  = (bf16_t*)carve(sizeof(bf16_t) * 64 * 128);
    bf16_t* w6b  = (bf16_t*)carve(sizeof(bf16_t) * 1024 * 192);
    bf16_t* wp1b = (bf16_t*)carve(sizeof(bf16_t) * 256 * 1280);
    bf16_t* wp2b = (bf16_t*)carve(sizeof(bf16_t) * 256 * 256);
    bf16_t* wp3b = (bf16_t*)carve(sizeof(bf16_t) * 128 * 256);
    (void)ws_size; (void)in_sizes; (void)n_in; (void)out_size;

    // ---- weight conversion (fp32 -> bf16, Ci zero-padded to %32) ----
    auto cvt = [&](const float* src, bf16_t* dst, int Co, int Ci, int CiPad) {
        const int tot = Co * CiPad;
        k_cvt_w<<<(tot + 255) / 256, 256, 0, stream>>>(src, dst, Co, Ci, CiPad);
    };
    cvt(w1,  w1b,  64,   6,    32);
    cvt(w2,  w2b,  64,   64,   64);
    cvt(w3,  w3b,  64,   128,  128);
    cvt(w4,  w4b,  64,   64,   64);
    cvt(w5,  w5b,  64,   128,  128);
    cvt(w6,  w6b,  1024, 192,  192);
    cvt(wp1, wp1b, 256,  1280, 1280);
    cvt(wp2, wp2b, 256,  256,  256);
    cvt(wp3, wp3b, 128,  256,  256);

    auto gemm = [&](const bf16_t* W, const bf16_t* X, const float* sc, const float* bi,
                    bf16_t* Y, int Co, int CiPad, int P, int act) {
        dim3 grid(P / 256, Co / 16);
        k_gemm_bn_act<<<grid, 128, 0, stream>>>(W, X, sc, bi, Y, Co, CiPad, P, act);
    };

    // ===== Block 1: knn on xyz, edge conv 6->64 -> 64->64, max over K =====
    k_knn3<<<(P1 + 255) / 256, 256, 0, stream>>>(points, idx);
    k_edge3<<<(PK + 255) / 256, 256, 0, stream>>>(points, idx, e_buf);
    gemm(w1b, e_buf, s1, b1, h_buf, 64, 32, PK, 1);
    gemm(w2b, h_buf, s2, b2, e_buf, 64, 64, PK, 1);        // reuse e_buf blocks 0-3
    k_maxk<<<(64 * P1 + 255) / 256, 256, 0, stream>>>(e_buf, X1280, 0);     // out1

    // ===== Block 2: knn on out1, edge conv 128->64 -> 64->64, max over K =====
    {
        dim3 g(NPTS / 64, BATCH);
        k_knn_feat<<<g, 64, 0, stream>>>(X1280, 0, idx);
    }
    k_edge_feat<<<(PK + 255) / 256, 256, 0, stream>>>(X1280, 0, idx, e_buf);
    gemm(w3b, e_buf, s3, b3, h_buf, 64, 128, PK, 1);
    gemm(w4b, h_buf, s4, b4, e_buf, 64, 64, PK, 1);
    k_maxk<<<(64 * P1 + 255) / 256, 256, 0, stream>>>(e_buf, X1280, 64);    // out2

    // ===== Block 3: knn on out2, edge conv 128->64, max over K =====
    {
        dim3 g(NPTS / 64, BATCH);
        k_knn_feat<<<g, 64, 0, stream>>>(X1280, 64, idx);
    }
    k_edge_feat<<<(PK + 255) / 256, 256, 0, stream>>>(X1280, 64, idx, e_buf);
    gemm(w5b, e_buf, s5, b5, h_buf, 64, 128, PK, 1);
    k_maxk<<<(64 * P1 + 255) / 256, 256, 0, stream>>>(h_buf, X1280, 128);   // out3

    // ===== Global feature: conv 192->1024 + max over N; label branch =====
    gemm(w6b, X1280 /* pack16 blocks 0-11 contiguous */, s6, b6, out4full, 1024, 192, P1, 1);
    k_maxn<<<(BATCH * 1024 + 255) / 256, 256, 0, stream>>>(out4full, out4max);
    k_label<<<(BATCH * 64 + 255) / 256, 256, 0, stream>>>(label, w7, s7, b7, catvet);
    k_glob<<<(1088 * P1 + 255) / 256, 256, 0, stream>>>(out4max, catvet, X1280);

    // ===== Segmentation head: 1280->256->256->128->6 =====
    gemm(wp1b, X1280, sp1, bp1, y256a, 256, 1280, P1, 1);
    gemm(wp2b, y256a, sp2, bp2, y256b, 256, 256, P1, 1);
    gemm(wp3b, y256b, sp3, bp3, y128,  128, 256, P1, 1);
    k_final<<<(P1 + 255) / 256, 256, 0, stream>>>(y128, wp4, out);
}
